// dyspatial_module_2688649527680
// MI455X (gfx1250) — matrix operand order repaired and checked by reading, not compiled, not run on hardware
//
#include <hip/hip_runtime.h>

#define N_IMG 16
#define CCH   256
#define HH    64
#define WWID  64
#define HP    66
#define WP    66
#define KTAP  9
#define KTOT  (KTAP*CCH)      // 2304
#define NPIX  (N_IMG*HH*WWID) // 65536
#define BN_EPS 1e-5f

// LDS row strides (bf16 elements) — data + TDM pad (32B) to break bank conflicts
#define SA_STRIDE 48    // 32 ch * 2B = 64B data + 32B pad = 96B
#define SB_STRIDE 272   // 256 k * 2B = 512B data + 32B pad = 544B

typedef __attribute__((ext_vector_type(16))) __bf16          v16bf;
typedef __attribute__((ext_vector_type(8)))  float           v8f;
typedef __attribute__((ext_vector_type(8)))  unsigned short  v8us;
typedef __attribute__((ext_vector_type(4)))  unsigned int    u32x4;
typedef __attribute__((ext_vector_type(8)))  int             i32x8;
typedef __attribute__((ext_vector_type(4)))  int             i32x4;

#if defined(__has_builtin)
#if __has_builtin(__builtin_amdgcn_tensor_load_to_lds) && __has_builtin(__builtin_amdgcn_s_wait_tensorcnt)
#define HAVE_TDM 1
#endif
#endif
#ifndef HAVE_TDM
#define HAVE_TDM 0
#endif

union Frag {
    v16bf v;
    v8us  h[2];
    unsigned short us[16];
};

__device__ __forceinline__ unsigned short f2bf(float f) {
    union { float f; unsigned u; } cv; cv.f = f;
    unsigned u = cv.u;
    u += 0x7FFFu + ((u >> 16) & 1u);        // round-to-nearest-even
    return (unsigned short)(u >> 16);
}
__device__ __forceinline__ float bf2f(unsigned short h) {
    union { float f; unsigned u; } cv; cv.u = ((unsigned)h) << 16; return cv.f;
}

#if HAVE_TDM
// ---------------------------------------------------------------------------
// Issue one TENSOR_LOAD_TO_LDS.  D# built per CDNA5 ISA.
// 16-bit elements, up to 3D tile, LDS padding enabled.
// ---------------------------------------------------------------------------
__device__ __forceinline__ void tdm_load(const unsigned short* gptr, unsigned lds_off,
                                         unsigned td0, unsigned ts0,   // tensor dim0 len/stride
                                         unsigned td1, unsigned ts1,   // tensor dim1 len/stride
                                         unsigned td2,                 // tensor dim2 len
                                         unsigned tl0, unsigned tl1, unsigned tl2,
                                         unsigned padi, unsigned pada) {
    unsigned long long ga = (unsigned long long)gptr;
    u32x4 g0;
    g0[0] = 1u;                                           // count=1, user descriptor
    g0[1] = lds_off;                                      // lds_addr
    g0[2] = (unsigned)ga;                                 // global_addr[31:0]
    g0[3] = (unsigned)((ga >> 32) & 0x01FFFFFFu) | (2u << 30);  // ga[56:32] | type=2
    i32x8 g1;
    g1[0] = (int)((1u << 16) | (1u << 20) | (padi << 22) | (pada << 25)); // data_size=2B, pad_enable
    g1[1] = (int)(td0 << 16);                             // tensor_dim0[15:0]
    g1[2] = (int)((td0 >> 16) | (td1 << 16));             // tensor_dim0[31:16] | tensor_dim1[15:0]
    g1[3] = (int)((td1 >> 16) | (tl0 << 16));             // tensor_dim1[31:16] | tile_dim0
    g1[4] = (int)((tl1 & 0xFFFFu) | (tl2 << 16));         // tile_dim1 | tile_dim2
    g1[5] = (int)ts0;                                     // tensor_dim0_stride[31:0]
    g1[6] = (int)((ts1 & 0xFFFFu) << 16);                 // ts0[47:32]=0 | ts1[15:0]
    g1[7] = (int)(ts1 >> 16);                             // ts1[47:16]
    i32x4 g2; g2[0] = (int)td2; g2[1] = 0; g2[2] = 0; g2[3] = 0;
    i32x4 g3; g3[0] = 0; g3[1] = 0; g3[2] = 0; g3[3] = 0;
#if __clang_major__ >= 23
    i32x8 gz; gz[0]=0; gz[1]=0; gz[2]=0; gz[3]=0; gz[4]=0; gz[5]=0; gz[6]=0; gz[7]=0;
    __builtin_amdgcn_tensor_load_to_lds(g0, g1, g2, g3, gz, 0);
#else
    __builtin_amdgcn_tensor_load_to_lds(g0, g1, g2, g3, 0);
#endif
}
#endif

// ---------------------------------------------------------------------------
// Pre-pass 0: zero the padded NHWC buffer (borders must read as 0)
// ---------------------------------------------------------------------------
__global__ void prep_zero(uint4* __restrict__ p, int n16) {
    int i = blockIdx.x * 256 + threadIdx.x;
    if (i < n16) p[i] = make_uint4(0u, 0u, 0u, 0u);
}

// ---------------------------------------------------------------------------
// Pre-pass 1: y fp32 NCHW -> bf16 padded NHWC  ypad2[n][hh][ww][c]
// (tiled 32x32 LDS transpose: coalesced reads over pixels, coalesced writes
//  over channels)
// ---------------------------------------------------------------------------
__global__ void prep_ypad_nhwc(const float* __restrict__ y,
                               unsigned short* __restrict__ ypad2) {
    __shared__ float s[32][33];
    const int bid = blockIdx.x;
    const int pt  = bid & 127;        // 128 pixel tiles of 32
    const int ct  = (bid >> 7) & 7;   // 8 channel tiles of 32
    const int n   = bid >> 10;
    const int p0  = pt * 32;
    const int c0  = ct * 32;
    const int tx  = threadIdx.x & 31;
    const int ty  = threadIdx.x >> 5; // 0..7
    const float* src = y + ((long long)(n * CCH + c0) << 12) + p0;
    #pragma unroll
    for (int i = 0; i < 4; ++i) {
        const int cr = ty + i * 8;
        s[cr][tx] = src[((long long)cr << 12) + tx];
    }
    __syncthreads();
    #pragma unroll
    for (int i = 0; i < 4; ++i) {
        const int pr  = ty + i * 8;
        const int pix = p0 + pr;
        const int hh  = (pix >> 6) + 1;
        const int ww  = (pix & 63) + 1;
        ypad2[(long long)((n * HP + hh) * WP + ww) * CCH + c0 + tx] = f2bf(s[tx][pr]);
    }
}

// ---------------------------------------------------------------------------
// Pre-pass 2: W1 [O][I][3][3] fp32 -> bf16 Bpack[o][k], k = tap*256 + ci
// ---------------------------------------------------------------------------
__global__ void prep_w(const float* __restrict__ W1,
                       unsigned short* __restrict__ bpack) {
    int i = blockIdx.x * 256 + threadIdx.x;
    if (i >= CCH * KTOT) return;
    int k = i % KTOT;
    int o = i / KTOT;
    int tap = k / CCH;
    int ci  = k % CCH;
    int kh = tap / 3, kw = tap % 3;
    bpack[i] = f2bf(W1[((o * CCH + ci) * 3 + kh) * 3 + kw]);
}

// ---------------------------------------------------------------------------
// Main: implicit-GEMM conv3x3, V_WMMA_F32_16X16X32_BF16, fused BN+PReLU.
// Tile: 128 pixels x 64 oc per workgroup.  A/B tiles DMA'd into LDS by the
// Tensor Data Mover (wave 0 issues, TENSORcnt-pipelined, double-buffered A),
// with TDM LDS padding for conflict-free fragment reads.
// ---------------------------------------------------------------------------
__launch_bounds__(256, 2)
__global__ void conv3x3_wmma(const unsigned short* __restrict__ ypad2,
                             const unsigned short* __restrict__ bpack,
                             const float* __restrict__ g1,
                             const float* __restrict__ b1,
                             const float* __restrict__ m1,
                             const float* __restrict__ v1,
                             const float* __restrict__ a1p,
                             unsigned short* __restrict__ hout) {
    __shared__ __align__(16) unsigned short sA[2][128 * SA_STRIDE]; // [pixel][32 ch] per buf
    __shared__ __align__(16) unsigned short sB[64 * SB_STRIDE];     // [oc][256 k of one tap]

    const int tid  = threadIdx.x;
    const int lane = tid & 31;
    const int wave = tid >> 5;
    const int wm   = wave & 3;   // M offset = wm*32
    const int wn   = wave >> 2;  // N offset = wn*32

    const int pb   = blockIdx.x >> 2;
    const int ob   = blockIdx.x & 3;
    const int pix0 = pb * 128;
    const int o0   = ob * 64;
    const int n    = pix0 >> 12;
    const int h0   = (pix0 & 4095) >> 6;   // block = 2 full image rows

    v8f acc[2][2] = {};

    const int mrow = lane & 15;
    const int kbA  = (lane >> 4) * 8;   // A frag: elems 0..7 -> K=kbA.., 8..15 -> K=16+kbA..
    const int kbB  = (lane >> 4) * 8;   // B frag: elems 0..7 -> K=kbB.., 8..15 -> K=16+kbB..

#if HAVE_TDM
    const unsigned lds_b    = (unsigned)(unsigned long long)(void*)sB;
    const unsigned lds_a[2] = { (unsigned)(unsigned long long)(void*)sA[0],
                                (unsigned)(unsigned long long)(void*)sA[1] };
#endif

    for (int tap = 0; tap < 9; ++tap) {
        const int kh = tap / 3, kw = tap % 3;
        const unsigned short* aBase =
            ypad2 + (long long)((n * HP + h0 + kh) * WP + kw) * CCH;

#if HAVE_TDM
        if (wave == 0) {
            // B panel: 64 oc x 256 k (2D), one DMA per tap
            tdm_load(bpack + (long long)o0 * KTOT + tap * CCH, lds_b,
                     KTOT, KTOT, CCH, 0, 0, CCH, 64, 0, /*padi*/6, /*pada*/7);
            // A chunk 0: [2 rows][64 cols][32 ch] 3D box
            tdm_load(aBase, lds_a[0],
                     CCH, CCH, WP, WP * CCH, HP, 32, 64, 2, /*padi*/3, /*pada*/7);
        }
#else
        {   // synchronous B staging fallback: 8 x b128 per thread
            const int o  = tid >> 2;
            const int kb = (tid & 3) * 64;
            const unsigned short* src = bpack + (long long)(o0 + o) * KTOT + tap * CCH + kb;
            #pragma unroll
            for (int j = 0; j < 8; ++j)
                *(v8us*)&sB[o * SB_STRIDE + kb + j * 8] = *(const v8us*)(src + j * 8);
        }
#endif

        for (int i = 0; i < 8; ++i) {
            const int c0 = i * 32;
#if HAVE_TDM
            if (wave == 0) {
                if (i < 7) {   // prefetch next A chunk into the other buffer
                    tdm_load(aBase + (i + 1) * 32, lds_a[(i + 1) & 1],
                             CCH, CCH, WP, WP * CCH, HP, 32, 64, 2, 3, 7);
                    __builtin_amdgcn_s_wait_tensorcnt(1);
                } else {
                    __builtin_amdgcn_s_wait_tensorcnt(0);
                }
            }
            const unsigned short* pA = sA[i & 1];
#else
            {   // synchronous A staging fallback: 1 x b128 per thread
                const int l    = tid >> 1;
                const int part = (tid & 1) * 16;
                const int hh   = h0 + kh + (l >> 6);
                const int ww   = kw + (l & 63);
                *(v8us*)&sA[0][l * SA_STRIDE + part] =
                    *(const v8us*)(ypad2 + (long long)((n * HP + hh) * WP + ww) * CCH
                                   + c0 + part);
            }
            const unsigned short* pA = sA[0];
#endif
            __syncthreads();

            Frag fa[2], fb[2];
            #pragma unroll
            for (int mi = 0; mi < 2; ++mi) {
                const int r = wm * 32 + mi * 16 + mrow;
                fa[mi].h[0] = *(const v8us*)&pA[r * SA_STRIDE + kbA];
                fa[mi].h[1] = *(const v8us*)&pA[r * SA_STRIDE + kbA + 16];
            }
            #pragma unroll
            for (int ni = 0; ni < 2; ++ni) {
                const int r = wn * 32 + ni * 16 + mrow;
                fb[ni].h[0] = *(const v8us*)&sB[r * SB_STRIDE + c0 + kbB];
                fb[ni].h[1] = *(const v8us*)&sB[r * SB_STRIDE + c0 + kbB + 16];
            }
            #pragma unroll
            for (int mi = 0; mi < 2; ++mi)
                #pragma unroll
                for (int ni = 0; ni < 2; ++ni)
                    acc[mi][ni] = __builtin_amdgcn_wmma_f32_16x16x32_bf16(
                        false, fa[mi].v, false, fb[ni].v,
                        (short)0, acc[mi][ni], false, false);
            __syncthreads();
        }
    }

    // epilogue: BN + PReLU, store bf16 h[pixel][oc]
    const float alpha = a1p[0];
    #pragma unroll
    for (int ni = 0; ni < 2; ++ni) {
        const int oc = o0 + wn * 32 + ni * 16 + (lane & 15);
        const float sc = g1[oc] * rsqrtf(v1[oc] + BN_EPS);
        const float sh = b1[oc] - m1[oc] * sc;
        #pragma unroll
        for (int mi = 0; mi < 2; ++mi) {
            #pragma unroll
            for (int r = 0; r < 8; ++r) {
                const int M  = r + ((lane >> 4) << 3);
                const int gp = pix0 + wm * 32 + mi * 16 + M;
                float val = acc[mi][ni][r];
                val = val * sc + sh;
                val = (val >= 0.f) ? val : alpha * val;
                hout[(long long)gp * CCH + oc] = f2bf(val);
            }
        }
    }
}

// ---------------------------------------------------------------------------
// conv1x1 256 -> 9 + BN + PReLU (tiny GEMV per pixel)
// ---------------------------------------------------------------------------
__global__ void conv1x1_kern(const unsigned short* __restrict__ hin,
                             const float* __restrict__ W2,
                             const float* __restrict__ g2,
                             const float* __restrict__ b2,
                             const float* __restrict__ m2,
                             const float* __restrict__ v2,
                             const float* __restrict__ a2p,
                             float* __restrict__ kern) {
    __shared__ float sW[KTAP][CCH];
    const int tid = threadIdx.x;
    #pragma unroll
    for (int j = 0; j < KTAP; ++j) sW[j][tid] = W2[j * CCH + tid];
    __syncthreads();

    const int p = blockIdx.x * 256 + tid;
    float acc[KTAP] = {};
    const unsigned short* hp = hin + (long long)p * CCH;
    for (int c = 0; c < CCH; ++c) {
        const float hv = bf2f(hp[c]);
        #pragma unroll
        for (int j = 0; j < KTAP; ++j) acc[j] = fmaf(hv, sW[j][c], acc[j]);
    }
    const float alpha = a2p[0];
    #pragma unroll
    for (int j = 0; j < KTAP; ++j) {
        const float sc = g2[j] * rsqrtf(v2[j] + BN_EPS);
        const float sh = b2[j] - m2[j] * sc;
        float v = acc[j] * sc + sh;
        kern[p * KTAP + j] = (v >= 0.f) ? v : alpha * v;
    }
}

// ---------------------------------------------------------------------------
// Apply per-pixel dynamic 3x3 filter depthwise over x (memory-bound)
// ---------------------------------------------------------------------------
__global__ void apply_dyn(const float* __restrict__ x,
                          const float* __restrict__ kern,
                          float* __restrict__ out) {
    const int idx = blockIdx.x * 256 + threadIdx.x;
    const int w = idx & 63;
    const int h = (idx >> 6) & 63;
    const int c = (idx >> 12) & 255;
    const int n = idx >> 20;
    const int p = (n << 12) | (h << 6) | w;
    const float* kp = kern + p * KTAP;
    const float* xp = x + ((long long)(n * CCH + c) << 12);
    float s = 0.f;
    #pragma unroll
    for (int dh = 0; dh < 3; ++dh) {
        const int hy = h + dh - 1;
        if ((unsigned)hy >= (unsigned)HH) continue;
        #pragma unroll
        for (int dw = 0; dw < 3; ++dw) {
            const int wx = w + dw - 1;
            if ((unsigned)wx >= (unsigned)WWID) continue;
            s = fmaf(xp[(hy << 6) + wx], kp[dh * 3 + dw], s);
        }
    }
    out[idx] = s;
}

// ---------------------------------------------------------------------------
extern "C" void kernel_launch(void* const* d_in, const int* in_sizes, int n_in,
                              void* d_out, int out_size, void* d_ws, size_t ws_size,
                              hipStream_t stream) {
    (void)in_sizes; (void)n_in; (void)out_size; (void)ws_size;

    const float* x  = (const float*)d_in[0];
    const float* y  = (const float*)d_in[1];
    const float* W1 = (const float*)d_in[2];
    const float* g1 = (const float*)d_in[3];
    const float* b1 = (const float*)d_in[4];
    const float* m1 = (const float*)d_in[5];
    const float* v1 = (const float*)d_in[6];
    const float* a1 = (const float*)d_in[7];
    const float* W2 = (const float*)d_in[8];
    const float* g2 = (const float*)d_in[9];
    const float* b2 = (const float*)d_in[10];
    const float* m2 = (const float*)d_in[11];
    const float* v2 = (const float*)d_in[12];
    const float* a2 = (const float*)d_in[13];
    float* out = (float*)d_out;

    char* ws = (char*)d_ws;
    unsigned short* ypad2 = (unsigned short*)(ws);              // 35,684,352 B
    unsigned short* bpack = (unsigned short*)(ws + 35684352);   //  1,179,648 B
    unsigned short* hbuf  = (unsigned short*)(ws + 36864000);   // 33,554,432 B
    float*          kern  = (float*)         (ws + 70418432);   //  2,359,296 B

    const int ypad_u16 = N_IMG * CCH * HP * WP;                 // 17,842,176
    prep_zero<<<(ypad_u16 / 8 + 255) / 256, 256, 0, stream>>>((uint4*)ypad2, ypad_u16 / 8);
    prep_ypad_nhwc<<<N_IMG * 8 * 128, 256, 0, stream>>>(y, ypad2);
    prep_w<<<(CCH * KTOT + 255) / 256, 256, 0, stream>>>(W1, bpack);

    conv3x3_wmma<<<(NPIX / 128) * (CCH / 64), 256, 0, stream>>>(
        ypad2, bpack, g1, b1, m1, v1, a1, hbuf);

    conv1x1_kern<<<NPIX / 256, 256, 0, stream>>>(
        hbuf, W2, g2, b2, m2, v2, a2, kern);

    apply_dyn<<<(N_IMG * CCH * HH * WWID) / 256, 256, 0, stream>>>(
        x, kern, out);
}
